// NAOCell_37434934952279
// MI455X (gfx1250) — matrix operand order repaired and checked by reading, not compiled, not run on hardware
//
#include <hip/hip_runtime.h>
#include <hip/hip_bf16.h>

typedef __bf16 bf16_t;
typedef __attribute__((ext_vector_type(16))) __bf16 v16bf;
typedef __attribute__((ext_vector_type(8)))  __bf16 v8bf;
typedef __attribute__((ext_vector_type(8)))  float  v8f;

#define TSTEPS 512
#define BATCH  64
#define NINP   512
#define NHID   512
#define NWG    32
#define NTHR   256

// ---------- helpers ----------
static __device__ __forceinline__ bf16_t f2bf(float f) {
  union { float f; unsigned u; } v; v.f = f;
  unsigned r = v.u + 0x7FFFu + ((v.u >> 16) & 1u);   // round-to-nearest-even
  unsigned short h = (unsigned short)(r >> 16);
  return __builtin_bit_cast(bf16_t, h);
}

static __device__ __forceinline__ v16bf cat16(v8bf lo, v8bf hi) {
  v16bf r;
#pragma unroll
  for (int e = 0; e < 8; ++e) { r[e] = lo[e]; r[e + 8] = hi[e]; }
  return r;
}

static __device__ __forceinline__ v8bf ld8(const bf16_t* p) { return *(const v8bf*)p; }

// A fragment: 16x32 bf16, row-major source. ISA layout: lane<16 holds K k0..k0+7
// and k0+16..k0+23; lane>=16 holds the +8 shifted halves.  hs = 8*(lane>>4).
static __device__ __forceinline__ v16bf afrag(const bf16_t* arow, int k0, int hs) {
  return cat16(ld8(arow + k0 + hs), ld8(arow + k0 + 16 + hs));
}
// B fragment: 32x16 bf16, source is B^T row-major [N][K]. ISA layout: lane = N,
// lane<16 holds K k0..k0+7 and k0+16..k0+23; lane>=16 holds the +8 shifted halves.
static __device__ __forceinline__ v16bf bfrag(const bf16_t* brow, int k0, int kh) {
  return cat16(ld8(brow + k0 + kh / 2), ld8(brow + k0 + 16 + kh / 2));
}

static __device__ __forceinline__ float sigm(float x) {
  return 1.0f / (1.0f + __expf(-x));
}

static __device__ __forceinline__ void gridbar(int* bar, int* phase) {
  __syncthreads();
  if (threadIdx.x == 0) {
    int ph = ++(*phase);
    __hip_atomic_fetch_add(bar, 1, __ATOMIC_RELEASE, __HIP_MEMORY_SCOPE_AGENT);
    while (__hip_atomic_load(bar, __ATOMIC_ACQUIRE, __HIP_MEMORY_SCOPE_AGENT) < NWG * ph)
      __builtin_amdgcn_s_sleep(2);
  }
  __syncthreads();
}

// ---------- prep: zero barrier, convert x to bf16, transpose+convert weights ----
__global__ void nao_prep(const float* __restrict__ x,
                         const float* __restrict__ W0, const float* __restrict__ Ws,
                         bf16_t* __restrict__ xbf,
                         bf16_t* __restrict__ w0t, bf16_t* __restrict__ wst,
                         int* __restrict__ bar) {
  long idx = (long)blockIdx.x * blockDim.x + threadIdx.x;
  long stride = (long)gridDim.x * blockDim.x;
  if (idx == 0) *bar = 0;
  const long nx = (long)TSTEPS * BATCH * NINP;   // straight convert
  for (long i = idx; i < nx; i += stride) xbf[i] = f2bf(x[i]);
  const long n0 = 1024L * 1024L;                 // W0T[n][k] = W0[k][n]
  for (long i = idx; i < n0; i += stride) {
    long n = i >> 10, k = i & 1023;
    w0t[i] = f2bf(W0[(k << 10) + n]);
  }
  const long n1 = 8L * 1024L * 512L;             // WsT[s][n][k] = Ws[s][k][n]
  for (long i = idx; i < n1; i += stride) {
    long s = i >> 19;
    long r = i & ((1L << 19) - 1);
    long n = r >> 9, k = r & 511;
    wst[i] = f2bf(Ws[(s << 19) + (k << 10) + n]);
  }
}

// ---------- persistent recurrent kernel ----------
__global__ __launch_bounds__(NTHR) void nao_persistent(
    const bf16_t* __restrict__ xbf,  // [T][B][NINP] bf16
    const float* __restrict__ h0,    // [1][B][NHID] f32
    const bf16_t* __restrict__ w0t,  // [1024][1024] bf16 (W0^T)
    const bf16_t* __restrict__ wst,  // [8][1024][512] bf16 (Ws^T)
    bf16_t* __restrict__ st,         // [5][B][NHID] bf16 state scratch (s0..s4)
    bf16_t* __restrict__ hbf,        // [B][NHID] bf16 hidden
    float* __restrict__ out,         // [T][B][NHID] + [1][B][NHID] f32
    int* __restrict__ bar) {
  __shared__ float ch[BATCH][32];        // gate cols 0..15, cand cols 16..31
  __shared__ float sloc[5][BATCH][16];   // f32 s0..s4, this WG's columns
  __shared__ float hloc[BATCH][16];      // f32 hidden, this WG's columns
  __shared__ int s_phase;

  const int tid  = threadIdx.x;
  const int wg   = blockIdx.x;           // owns state cols [16*wg, 16*wg+16)
  const int lane = tid & 31;
  const int wave = tid >> 5;             // 0..7
  const int m0   = (wave & 3) * 16;      // M tile base
  const int half = wave >> 2;            // 0 = gate, 1 = candidate
  const int r15  = lane & 15;
  const int hs   = (lane >> 4) * 8;      // A half-select / D row offset
  const int kh   = (lane >> 4) * 16;     // B half-select
  const int col0 = wg * 16;
  const int n0   = half ? (NHID + col0) : col0;

  // each thread owns 4 fixed elements of the [64][16] column block
  const int e0   = tid * 4;
  const int row0 = e0 >> 4;
  const int c0   = e0 & 15;

  const int kPred[8] = {0, 0, 1, 1, 2, 2, 3, 4};

  if (tid == 0) s_phase = 0;
  __syncthreads();

  // load initial hidden (f32 local + bf16 global)
#pragma unroll
  for (int j = 0; j < 4; ++j) {
    float hv = h0[row0 * NHID + col0 + c0 + j];
    hloc[row0][c0 + j] = hv;
    hbf[row0 * NHID + col0 + c0 + j] = f2bf(hv);
  }
  gridbar(bar, &s_phase);

  for (int t = 0; t < TSTEPS; ++t) {
    // Launder a zero OFFSET (not the pointers): keeps pointer provenance so the
    // compiler still emits global_load (not flat_load), but makes every weight
    // address non-invariant w.r.t. the t-loop so panel loads can't be hoisted
    // out of the timestep loop (which previously caused scratch spills).
    size_t lz = 0;
    asm volatile("" : "+s"(lz));
    const bf16_t* w0t_p = w0t + lz;
    const bf16_t* wst_p = wst + lz;

    // ---- GEMM0: ch0 = [x_t | h] @ W0, this WG's paired (gate,cand) columns ----
    {
      const bf16_t* brow = w0t_p + (size_t)(n0 + r15) * 1024;
      const bf16_t* xrow = xbf + ((size_t)t * BATCH + m0 + r15) * NINP;
      const bf16_t* hrow = hbf + (size_t)(m0 + r15) * NHID;
      v8f acc = {};
#pragma unroll 4
      for (int k0 = 0; k0 < NINP; k0 += 32) {
        v16bf a = afrag(xrow, k0, hs);
        v16bf b = bfrag(brow, k0, kh);
        acc = __builtin_amdgcn_wmma_f32_16x16x32_bf16(false, a, false, b,
                                                      (short)0, acc, false, false);
      }
#pragma unroll 4
      for (int k0 = 0; k0 < NHID; k0 += 32) {
        v16bf a = afrag(hrow, k0, hs);
        v16bf b = bfrag(brow, NINP + k0, kh);
        acc = __builtin_amdgcn_wmma_f32_16x16x32_bf16(false, a, false, b,
                                                      (short)0, acc, false, false);
      }
#pragma unroll
      for (int v = 0; v < 8; ++v)
        ch[m0 + v + hs][half * 16 + r15] = acc[v];
    }
    __syncthreads();

    // s0 = h + sigmoid(ch_g) * (tanh(ch_c) - h)
#pragma unroll
    for (int j = 0; j < 4; ++j) {
      int cc = c0 + j;
      float cg = sigm(ch[row0][cc]);
      float hv = tanhf(ch[row0][16 + cc]);
      float sp = hloc[row0][cc];
      float sn = sp + cg * (hv - sp);
      sloc[0][row0][cc] = sn;
      st[(size_t)row0 * NHID + col0 + cc] = f2bf(sn);
    }
    float sum[4] = {0.f, 0.f, 0.f, 0.f};
    gridbar(bar, &s_phase);   // s0 visible to all WGs

    // ---- 8 arch steps (fully unrolled: pred/act become constants) ----
#pragma unroll
    for (int stp = 0; stp < 8; ++stp) {
      const int pred = kPred[stp];
      const int act  = stp & 3;   // tanh, relu, sigmoid, identity
      const bf16_t* arow = st + (size_t)pred * BATCH * NHID + (size_t)(m0 + r15) * NHID;
      const bf16_t* brow = wst_p + (size_t)stp * 1024 * 512 + (size_t)(n0 + r15) * 512;
      if (stp < 7) {  // prefetch next step's weight panel row (global_prefetch_b8)
        __builtin_prefetch((const void*)(wst_p + (size_t)(stp + 1) * 1024 * 512 +
                                         (size_t)(n0 + r15) * 512), 0, 1);
      }
      v8f acc = {};
#pragma unroll 4
      for (int k0 = 0; k0 < NHID; k0 += 32) {
        v16bf a = afrag(arow, k0, hs);
        v16bf b = bfrag(brow, k0, kh);
        acc = __builtin_amdgcn_wmma_f32_16x16x32_bf16(false, a, false, b,
                                                      (short)0, acc, false, false);
      }
#pragma unroll
      for (int v = 0; v < 8; ++v)
        ch[m0 + v + hs][half * 16 + r15] = acc[v];
      __syncthreads();

#pragma unroll
      for (int j = 0; j < 4; ++j) {
        int cc = c0 + j;
        float cg  = sigm(ch[row0][cc]);
        float pre = ch[row0][16 + cc];
        float hv;
        if      (act == 0) hv = tanhf(pre);
        else if (act == 1) hv = fmaxf(pre, 0.f);
        else if (act == 2) hv = sigm(pre);
        else               hv = pre;
        float sp = sloc[pred][row0][cc];
        float sn = sp + cg * (hv - sp);
        sum[j] += sn;
        if (stp < 4) {  // s1..s4 feed later GEMMs
          sloc[stp + 1][row0][cc] = sn;
          st[(size_t)(stp + 1) * BATCH * NHID + (size_t)row0 * NHID + col0 + cc] = f2bf(sn);
        }
      }
      if (stp < 4) gridbar(bar, &s_phase);   // new predecessor state visible
      else         __syncthreads();          // just protect ch reuse
    }

    // ---- finalize: h_new = mean(s1..s8), emit output, carry hidden ----
#pragma unroll
    for (int j = 0; j < 4; ++j) {
      int cc = c0 + j;
      float hv = sum[j] * 0.125f;
      hloc[row0][cc] = hv;
      hbf[(size_t)row0 * NHID + col0 + cc] = f2bf(hv);
      out[((size_t)t * BATCH + row0) * NHID + col0 + cc] = hv;
      if (t == TSTEPS - 1)
        out[(size_t)TSTEPS * BATCH * NHID + (size_t)row0 * NHID + col0 + cc] = hv;
    }
    gridbar(bar, &s_phase);  // hidden + state-buffer reuse safe for next t
  }
}

extern "C" void kernel_launch(void* const* d_in, const int* in_sizes, int n_in,
                              void* d_out, int out_size, void* d_ws, size_t ws_size,
                              hipStream_t stream) {
  const float* x  = (const float*)d_in[0];   // [512,64,512]
  const float* h0 = (const float*)d_in[1];   // [1,64,512]
  const float* W0 = (const float*)d_in[2];   // [1024,1024]
  const float* Ws = (const float*)d_in[3];   // [8,512,1024]
  char* ws = (char*)d_ws;
  int*    bar = (int*)ws;
  bf16_t* w0t = (bf16_t*)(ws + 256);
  bf16_t* wst = (bf16_t*)(ws + 256 + 2097152);
  bf16_t* st  = (bf16_t*)(ws + 256 + 2097152 + 8388608);
  bf16_t* hbf = (bf16_t*)(ws + 256 + 2097152 + 8388608 + 327680);
  bf16_t* xbf = (bf16_t*)(ws + 256 + 2097152 + 8388608 + 327680 + 65536);

  nao_prep<<<8192, 256, 0, stream>>>(x, W0, Ws, xbf, w0t, wst, bar);
  nao_persistent<<<NWG, NTHR, 0, stream>>>(xbf, h0, w0t, wst, st, hbf,
                                           (float*)d_out, bar);
}
